// StockPredictor_62697932587251
// MI455X (gfx1250) — matrix operand order repaired and checked by reading, not compiled, not run on hardware
//
#include <hip/hip_runtime.h>
#include <hip/hip_bf16.h>
#include <stdint.h>

// Problem sizes (match reference)
#define B_ 256
#define T_ 512
#define F_ 256
#define H_ 256
#define HSTRIDE 264           // padded LDS row stride (bf16 elems): 528B, 16B aligned, bank-friendly
#define MATELEMS (1024 * 256) // one transposed weight matrix [4H][K] in elements

typedef __attribute__((ext_vector_type(16))) __bf16 bf16x16;
typedef __attribute__((ext_vector_type(8)))  float  v8f;

union Frag { bf16x16 v; uint4 q[2]; };

__device__ __forceinline__ unsigned pk_bf16(float a, float b) {
  unsigned ua = __float_as_uint(a), ub = __float_as_uint(b);
  ua = (ua + 0x7FFFu + ((ua >> 16) & 1u)) >> 16;   // round-to-nearest-even
  ub = (ub + 0x7FFFu + ((ub >> 16) & 1u)) >> 16;
  return ua | (ub << 16);
}
__device__ __forceinline__ unsigned short bf16b(float a) {
  unsigned ua = __float_as_uint(a);
  return (unsigned short)((ua + 0x7FFFu + ((ua >> 16) & 1u)) >> 16);
}
__device__ __forceinline__ float sigm_(float x) { return 1.0f / (1.0f + __expf(-x)); }
__device__ __forceinline__ float tanh_(float x) {
  float a = fabsf(x);
  float e = __expf(-2.0f * a);
  float t = (1.0f - e) / (1.0f + e);
  return copysignf(t, x);
}
__device__ __forceinline__ v8f vzero() { v8f z = {0.f,0.f,0.f,0.f,0.f,0.f,0.f,0.f}; return z; }

#define WMMA_BF16(ACC, A, BF) \
  (ACC) = __builtin_amdgcn_wmma_f32_16x16x32_bf16(false, (A).v, false, (BF).v, (short)0, (ACC), false, false)

// Load the 4 gate B-fragments for one k-step as one burst (single clause/wait),
// then issue the 4 WMMAs. MOFF/g/kk offsets are compile-time -> load immediates.
#define GATE_STEP(ACC, A, WADR, EOFF)                                              \
  do {                                                                             \
    Frag B0, B1, B2, B3;                                                           \
    const unsigned short* pb_ = (WADR) + (EOFF);                                   \
    B0.q[0] = *(const uint4*)(pb_);               B0.q[1] = *(const uint4*)(pb_ + 16);              \
    B1.q[0] = *(const uint4*)(pb_ + 65536);       B1.q[1] = *(const uint4*)(pb_ + 65536 + 16);      \
    B2.q[0] = *(const uint4*)(pb_ + 131072);      B2.q[1] = *(const uint4*)(pb_ + 131072 + 16);     \
    B3.q[0] = *(const uint4*)(pb_ + 196608);      B3.q[1] = *(const uint4*)(pb_ + 196608 + 16);     \
    WMMA_BF16((ACC)[0], (A), B0);                                                  \
    WMMA_BF16((ACC)[1], (A), B1);                                                  \
    WMMA_BF16((ACC)[2], (A), B2);                                                  \
    WMMA_BF16((ACC)[3], (A), B3);                                                  \
  } while (0)

// K=256 GEMM slice, A from LDS hidden state.
template <int MOFF>
__device__ __forceinline__ void gemm_lds(v8f acc[4], const unsigned short* hs_lane,
                                         const unsigned short* wadr) {
#pragma unroll
  for (int kk = 0; kk < 8; ++kk) {
    Frag A;
    A.q[0] = *(const uint4*)(hs_lane + kk * 32);       // K = k0+kbA    .. +8
    A.q[1] = *(const uint4*)(hs_lane + kk * 32 + 16);  // K = k0+16+kbA .. +8
    GATE_STEP(acc, A, wadr, MOFF + kk * 32);
  }
}

// K=256 GEMM slice, A from pre-swizzled global bf16 x fragments.
template <int MOFF>
__device__ __forceinline__ void gemm_glb(v8f acc[4], const unsigned short* xa_lane,
                                         const unsigned short* wadr) {
#pragma unroll
  for (int kk = 0; kk < 8; ++kk) {
    Frag A;
    A.q[0] = *(const uint4*)(xa_lane + kk * 512);      // fragment stride = 32 lanes * 16 elems
    A.q[1] = *(const uint4*)(xa_lane + kk * 512 + 8);
    GATE_STEP(acc, A, wadr, MOFF + kk * 32);
  }
}

// Gate nonlinearities + cell update; cell state stays in VGPRs, h goes to LDS (bf16)
__device__ __forceinline__ void gate_update(v8f acc[4], v8f& c,
                                            float bi, float bff, float bg, float bo,
                                            unsigned short* hs, float* hf, bool wf32,
                                            int hi, int hbase, int n) {
#pragma unroll
  for (int v = 0; v < 8; ++v) {
    float ig = sigm_(acc[0][v] + bi);
    float fg = sigm_(acc[1][v] + bff);
    float gg = tanh_(acc[2][v] + bg);
    float og = sigm_(acc[3][v] + bo);
    float cn = fg * c[v] + ig * gg;
    c[v] = cn;
    float hv = og * tanh_(cn);
    int M = v + hi * 8;                         // C/D layout: row = v + 8*(lane>=16)
    hs[M * HSTRIDE + hbase + n] = bf16b(hv);
    if (wf32) hf[M * H_ + hbase + n] = hv;
  }
}

// ---- Pre-pass 1: convert + transpose 4 weight matrices f32[256][1024] -> bf16[1024][256]
__global__ __launch_bounds__(256)
void cvt_transpose_kernel(const float* __restrict__ W1, const float* __restrict__ U1,
                          const float* __restrict__ W2, const float* __restrict__ U2,
                          unsigned short* __restrict__ out) {
  int idx = blockIdx.x * 256 + threadIdx.x;
  int mat = idx >> 18;              // 262144 elems per matrix
  int r   = idx & (MATELEMS - 1);
  int g   = r >> 8;                 // gate column 0..1023
  int k   = r & 255;                // K index
  const float* src = (mat == 0) ? W1 : (mat == 1) ? U1 : (mat == 2) ? W2 : U2;
  out[idx] = bf16b(src[k * 1024 + g]);
}

// ---- Pre-pass 2: convert x f32[B][T][F] -> bf16 pre-swizzled into per-lane WMMA
// A-fragment order: xa[(((bt*T + t)*8 + kk)*32 + lane)*16 .. +16].
// Per the 16-bit A layout: lane holds M=lane%16, kb=(lane<16?0:8),
// elems 0..7 = K kk*32+kb+0..7, elems 8..15 = K kk*32+16+kb+0..7.
__global__ __launch_bounds__(256)
void xswizzle_kernel(const float* __restrict__ x, unsigned short* __restrict__ xa) {
  int tid  = blockIdx.x * 256 + threadIdx.x;    // 0 .. 2^21
  int lane = tid & 31;
  int kk   = (tid >> 5) & 7;
  int t    = (tid >> 8) & (T_ - 1);
  int bt   = tid >> 17;                         // 0..15
  int b    = bt * 16 + (lane & 15);
  int kb   = (lane < 16) ? 0 : 8;
  const float* px = x + ((size_t)b * T_ + t) * F_ + kk * 32 + kb;
  float4 f0 = *(const float4*)px,        f1 = *(const float4*)(px + 4);
  float4 f2 = *(const float4*)(px + 16), f3 = *(const float4*)(px + 20);
  uint4 q0, q1;
  q0.x = pk_bf16(f0.x, f0.y); q0.y = pk_bf16(f0.z, f0.w);
  q0.z = pk_bf16(f1.x, f1.y); q0.w = pk_bf16(f1.z, f1.w);
  q1.x = pk_bf16(f2.x, f2.y); q1.y = pk_bf16(f2.z, f2.w);
  q1.z = pk_bf16(f3.x, f3.y); q1.w = pk_bf16(f3.z, f3.w);
  uint4* po = (uint4*)(xa + (size_t)tid * 16);
  po[0] = q0;
  po[1] = q1;
}

// ---- Main scan: one workgroup per 16-batch tile; 16 waves; wave w owns hidden
// cols [16w,16w+16) of BOTH layers (its 4 gate tiles i/f/g/o are its own accums).
__global__ __launch_bounds__(512)
void lstm2_wmma_kernel(const unsigned short* __restrict__ xa,
                       const unsigned short* __restrict__ wbf,   // Wt1|Ut1|Wt2|Ut2
                       const float* __restrict__ b1,
                       const float* __restrict__ b2,
                       const float* __restrict__ Wd,
                       const float* __restrict__ bd,
                       float* __restrict__ out) {
  __shared__ __align__(16) unsigned short h1s[16 * HSTRIDE];
  __shared__ __align__(16) unsigned short h2s[16 * HSTRIDE];
  __shared__ __align__(16) float          h2f[16 * H_];

  const int tid   = threadIdx.x;
  const int lane  = tid & 31;
  const int wave  = tid >> 5;        // 0..15
  const int n     = lane & 15;       // A: batch row within tile; B/C/D: column
  const int hi    = lane >> 4;       // half-wave select
  const int kbA   = hi * 8;          // A-frag K base (16-bit A layout)
  const int kbB   = hi * 8;          // B-frag K base (16-bit B layout)
  const int btile = blockIdx.x;      // 0..15
  const int hbase = wave * 16;

  for (int i = tid; i < 16 * HSTRIDE; i += 512) { h1s[i] = 0; h2s[i] = 0; }
  __syncthreads();

  // Per-lane gate biases (constant over time): gate col = g*H + hbase + n
  const float bi1 = b1[0 * H_ + hbase + n], bf1 = b1[1 * H_ + hbase + n];
  const float bg1 = b1[2 * H_ + hbase + n], bo1 = b1[3 * H_ + hbase + n];
  const float bi2 = b2[0 * H_ + hbase + n], bf2 = b2[1 * H_ + hbase + n];
  const float bg2 = b2[2 * H_ + hbase + n], bo2 = b2[3 * H_ + hbase + n];

  // Single per-lane base pointers; matrix/gate/k offsets fold into load immediates.
  const unsigned short* wadr    = wbf + (size_t)(hbase + n) * 256 + kbB;
  const unsigned short* h1_lane = h1s + n * HSTRIDE + kbA;
  const unsigned short* h2_lane = h2s + n * HSTRIDE + kbA;
  const unsigned short* xa_lane = xa + (size_t)btile * (T_ * 4096) + lane * 16;

  v8f c1 = vzero(), c2 = vzero();

#pragma unroll 1
  for (int t = 0; t < T_; ++t) {
    // Compiler barrier: prevents LICM from hoisting the (otherwise provably
    // loop-invariant) weight-fragment loads out of the loop and spilling them
    // to scratch. Weights must be re-streamed from L2 each timestep.
    asm volatile("" ::: "memory");

    // Prefetch next timestep's x fragments (speculative; tail overrun dropped)
    __builtin_prefetch(xa_lane + 4096, 0, 0);

    // ---------------- layer 1: z1 = x_t @ W1 + h1 @ U1 + b1 ----------------
    v8f acc[4] = { vzero(), vzero(), vzero(), vzero() };
    gemm_glb<0 * MATELEMS>(acc, xa_lane, wadr);            // Wt1
    gemm_lds<1 * MATELEMS>(acc, h1_lane, wadr);            // Ut1
    __syncthreads();                                       // all reads of h1(t-1) done
    gate_update(acc, c1, bi1, bf1, bg1, bo1, h1s, h2f, false, hi, hbase, n);
    __syncthreads();                                       // h1(t) visible

    // ---------------- layer 2: z2 = h1_t @ W2 + h2 @ U2 + b2 ----------------
    acc[0] = vzero(); acc[1] = vzero(); acc[2] = vzero(); acc[3] = vzero();
    gemm_lds<2 * MATELEMS>(acc, h1_lane, wadr);            // Wt2
    gemm_lds<3 * MATELEMS>(acc, h2_lane, wadr);            // Ut2
    __syncthreads();                                       // all reads of h2(t-1) done
    gate_update(acc, c2, bi2, bf2, bg2, bo2, h2s, h2f, (t == T_ - 1), hi, hbase, n);
    __syncthreads();                                       // h2(t) visible

    xa_lane += 4096;                                       // next timestep's fragments
  }

  // ---------------- dense head: out[b] = h2[b,:] . Wd + bd ----------------
  float part = 0.f;
  const float* hb = h2f + wave * H_;                       // wave w -> batch row w
#pragma unroll
  for (int k = lane; k < H_; k += 32) part += hb[k] * Wd[k];
#pragma unroll
  for (int off = 16; off; off >>= 1) part += __shfl_down(part, off, 32);
  if (lane == 0) out[btile * 16 + wave] = part + bd[0];
}

extern "C" void kernel_launch(void* const* d_in, const int* in_sizes, int n_in,
                              void* d_out, int out_size, void* d_ws, size_t ws_size,
                              hipStream_t stream) {
  const float* x  = (const float*)d_in[0];
  const float* W1 = (const float*)d_in[1];
  const float* U1 = (const float*)d_in[2];
  const float* b1 = (const float*)d_in[3];
  const float* W2 = (const float*)d_in[4];
  const float* U2 = (const float*)d_in[5];
  const float* b2 = (const float*)d_in[6];
  const float* Wd = (const float*)d_in[7];
  const float* bd = (const float*)d_in[8];

  unsigned short* wbf = (unsigned short*)d_ws;             // 4 x 512KB transposed bf16 weights
  unsigned short* xa  = wbf + 4 * (size_t)MATELEMS;        // 64MB pre-swizzled bf16 x fragments

  cvt_transpose_kernel<<<(4 * MATELEMS) / 256, 256, 0, stream>>>(W1, U1, W2, U2, wbf);
  xswizzle_kernel<<<(B_ / 16) * T_ * 8 * 32 / 256, 256, 0, stream>>>(x, xa);
  lstm2_wmma_kernel<<<B_ / 16, 512, 0, stream>>>(xa, wbf, b1, b2, Wd, bd, (float*)d_out);
}
